// Fusion_R3D_86371792323176
// MI455X (gfx1250) — matrix operand order repaired and checked by reading, not compiled, not run on hardware
//
#include <hip/hip_runtime.h>
#include <hip/hip_bf16.h>

// Problem sizes
#define BATCH   128
#define KDIM    8192
#define NDIM    2048
#define NCOLS   6144      // 3 * NDIM
#define NCLS    18

// GEMM tiling
#define BN      32        // N per block
#define BK      32        // K step
#define LDA     40        // padded LDS row length in bf16 elems (80B -> 20-bank stride, conflict free)

typedef __attribute__((ext_vector_type(4)))  float  f4;
typedef __attribute__((ext_vector_type(8)))  float  v8f;
typedef __attribute__((ext_vector_type(4)))  __bf16 v4bf;
typedef __attribute__((ext_vector_type(8)))  __bf16 v8bf;
typedef __attribute__((ext_vector_type(16))) __bf16 v16bf;

__device__ __forceinline__ v4bf cvt4(f4 v) {
  // native f32->bf16 converts (v_cvt_pk_bf16_f32 path), RNE
  v4bf r;
  r.x = (__bf16)v.x; r.y = (__bf16)v.y; r.z = (__bf16)v.z; r.w = (__bf16)v.w;
  return r;
}

// ---------------------------------------------------------------------------
// Kernel 0: pre-convert x (3 x [128,8192] f32) -> bf16 in workspace.
// Removes all A-side conversion VALU from the GEMM hot loop.
// ---------------------------------------------------------------------------
__global__ __launch_bounds__(256)
void xcvt_kernel(const float* __restrict__ X0, const float* __restrict__ X1,
                 const float* __restrict__ X2, __bf16* __restrict__ Xb) {
  const size_t i = ((size_t)blockIdx.x * 256 + threadIdx.x) * 8;  // 8 elems/thread
  const int branch = (int)(i >> 20);                  // 128*8192 = 2^20 per branch
  const size_t off = i & ((1u << 20) - 1);
  const float* X = (branch == 0) ? X0 : ((branch == 1) ? X1 : X2);
  f4 v0 = *(const f4*)(X + off);
  f4 v1 = *(const f4*)(X + off + 4);
  *(v4bf*)(Xb + i)     = cvt4(v0);
  *(v4bf*)(Xb + i + 4) = cvt4(v1);
}

// ---------------------------------------------------------------------------
// Kernel 1: wsum[branch][k] = sum_j Wan[k, j]   (row sums, one wave per row)
// ---------------------------------------------------------------------------
__global__ __launch_bounds__(256)
void rowsum_kernel(const float* __restrict__ W0, const float* __restrict__ W1,
                   const float* __restrict__ W2, float* __restrict__ wsum) {
  const int wave = threadIdx.x >> 5;
  const int lane = threadIdx.x & 31;
  const int row  = blockIdx.x * 8 + wave;         // 0 .. 3*8192-1
  const int branch = row >> 13;
  const int k      = row & (KDIM - 1);
  const float* W = (branch == 0) ? W0 : ((branch == 1) ? W1 : W2);
  const float* p = W + (size_t)k * NDIM;

  float sum = 0.f;
#pragma unroll
  for (int i = 0; i < 16; ++i) {                  // 2048 / (32 lanes * 4) = 16
    f4 v = __builtin_nontemporal_load((const f4*)(p + lane * 4 + i * 128));
    sum += v.x + v.y + v.z + v.w;
  }
#pragma unroll
  for (int off = 16; off > 0; off >>= 1) sum += __shfl_xor(sum, off, 32);
  if (lane == 0) wsum[row] = sum;
}

// ---------------------------------------------------------------------------
// Kernel 2: S[branch][b] = (1/F) * dot(x[b,:], wsum[branch,:])  (wave per dot)
// ---------------------------------------------------------------------------
__global__ __launch_bounds__(256)
void scale_kernel(const float* __restrict__ X0, const float* __restrict__ X1,
                  const float* __restrict__ X2, const float* __restrict__ wsum,
                  float* __restrict__ S) {
  const int wave = threadIdx.x >> 5;
  const int lane = threadIdx.x & 31;
  const int idx  = blockIdx.x * 8 + wave;         // 0 .. 383
  const int branch = idx >> 7;
  const int b      = idx & (BATCH - 1);
  const float* X  = (branch == 0) ? X0 : ((branch == 1) ? X1 : X2);
  const float* xp = X + (size_t)b * KDIM;
  const float* wp = wsum + branch * KDIM;

  float dot = 0.f;
#pragma unroll 8
  for (int i = 0; i < 64; ++i) {                  // 8192 / 128
    f4 xv = *(const f4*)(xp + lane * 4 + i * 128);
    f4 wv = *(const f4*)(wp + lane * 4 + i * 128);
    dot += xv.x * wv.x + xv.y * wv.y + xv.z * wv.z + xv.w * wv.w;
  }
#pragma unroll
  for (int off = 16; off > 0; off >>= 1) dot += __shfl_xor(dot, off, 32);
  if (lane == 0) S[idx] = dot * (1.0f / (float)NDIM);
}

// ---------------------------------------------------------------------------
// Kernel 3: V[b, colOff+n] = S[b] * (xbf @ bf16(W))[b, n]  via bf16 WMMA.
//   Block: 256 thr (8 waves). Tile M=128 x N=32, K-step 32. Wave w -> rows
//   [16w,16w+16), two 16x16 accumulators (n0 = 0, 16). A staged as a pure
//   bf16 copy (pre-converted); only B (the 192MB HBM stream) converts inline.
// ---------------------------------------------------------------------------
__global__ __launch_bounds__(256)
void gemm_scale_kernel(const __bf16* __restrict__ Xb,  // [128, 8192] bf16
                       const float*  __restrict__ W,   // [8192, 2048] f32
                       const float*  __restrict__ S,   // [128]
                       float*        __restrict__ V,   // [128, 6144]
                       int colOff) {
  __shared__ __bf16 As[BATCH * LDA];  // 10240 B, row-major [m][k]
  __shared__ __bf16 Bs[BN * LDA];     //  2560 B, transposed [n][k]

  const int t    = threadIdx.x;
  const int wave = t >> 5;
  const int lane = t & 31;
  const int hi   = lane >> 4;                 // 0: lanes 0-15, 1: lanes 16-31
  const int lm   = lane & 15;
  const int nBase = blockIdx.x * BN;
  const int mTile = wave * 16;

  // staging assignment
  const int aRow = t >> 2;                    // 0..63  (A: v8bf each, 2 rows)
  const int aCol = (t & 3) * 8;               // 0,8,16,24
  const int bRow = t >> 3;                    // 0..31  (B: one f4)
  const int bCol = (t & 7) * 4;               // 0..28

  v8f acc0 = {}; v8f acc1 = {};
  v8bf ar0, ar1;
  f4 br;

  // preload tile 0
  {
    const __bf16* xp = Xb + (size_t)aRow * KDIM + aCol;
    ar0 = *(const v8bf*)(xp);
    ar1 = *(const v8bf*)(xp + 64 * KDIM);
    br  = __builtin_nontemporal_load(
        (const f4*)(W + (size_t)bRow * NDIM + nBase + bCol));
  }

  for (int k0 = 0; k0 < KDIM; k0 += BK) {
    __syncthreads();                          // prior compute done reading LDS
    // stage registers -> LDS
    {
      *(v8bf*)&As[(aRow     ) * LDA + aCol] = ar0;
      *(v8bf*)&As[(aRow + 64) * LDA + aCol] = ar1;
      v4bf bb = cvt4(br);                     // native bf16 converts
      Bs[(bCol + 0) * LDA + bRow] = bb.x;     // transpose into [n][k]
      Bs[(bCol + 1) * LDA + bRow] = bb.y;
      Bs[(bCol + 2) * LDA + bRow] = bb.z;
      Bs[(bCol + 3) * LDA + bRow] = bb.w;
    }
    __syncthreads();

    // prefetch next tile into registers (overlaps with WMMA below)
    const int kn = k0 + BK;
    if (kn < KDIM) {
      const __bf16* xp = Xb + (size_t)aRow * KDIM + kn + aCol;
      ar0 = *(const v8bf*)(xp);
      ar1 = *(const v8bf*)(xp + 64 * KDIM);
      br  = __builtin_nontemporal_load(
          (const f4*)(W + (size_t)(kn + bRow) * NDIM + nBase + bCol));
      if (kn + BK < KDIM)
        __builtin_prefetch(W + (size_t)(kn + BK + bRow) * NDIM + nBase + bCol, 0, 1);
    }

    // A fragment: lane<16 -> K 0..7 & 16..23, lane>=16 -> K 8..15 & 24..31
    const __bf16* ap = &As[(mTile + lm) * LDA + hi * 8];
    v8bf alo = *(const v8bf*)(ap);
    v8bf ahi = *(const v8bf*)(ap + 16);
    v16bf a;
#pragma unroll
    for (int i = 0; i < 8; ++i) { a[i] = alo[i]; a[i + 8] = ahi[i]; }

    // B fragments: lane<16 -> col lm, K 0..7 & 16..23; lane>=16 -> col lm, K 8..15 & 24..31
    const __bf16* bp0 = &Bs[lm * LDA + hi * 8];
    v8bf b0l = *(const v8bf*)(bp0);
    v8bf b0h = *(const v8bf*)(bp0 + 16);
    const __bf16* bp1 = &Bs[(16 + lm) * LDA + hi * 8];
    v8bf b1l = *(const v8bf*)(bp1);
    v8bf b1h = *(const v8bf*)(bp1 + 16);
    v16bf b0, b1;
#pragma unroll
    for (int i = 0; i < 8; ++i) {
      b0[i] = b0l[i]; b0[i + 8] = b0h[i];
      b1[i] = b1l[i]; b1[i + 8] = b1h[i];
    }

    acc0 = __builtin_amdgcn_wmma_f32_16x16x32_bf16(
        false, a, false, b0, (short)0, acc0, false, false);
    acc1 = __builtin_amdgcn_wmma_f32_16x16x32_bf16(
        false, a, false, b1, (short)0, acc1, false, false);
  }

  // epilogue: VGPR r holds M = mTile + 8*hi + r, N = lm (+n0); fuse scale
#pragma unroll
  for (int r = 0; r < 8; ++r) {
    const int m = mTile + hi * 8 + r;
    const float sv = S[m];
    V[(size_t)m * NCOLS + colOff + nBase + lm]      = sv * acc0[r];
    V[(size_t)m * NCOLS + colOff + nBase + 16 + lm] = sv * acc1[r];
  }
}

// ---------------------------------------------------------------------------
// Kernel 4: width-3 avg pool over [B,6144] -> [B,2048], classifier + softmax
//   One block per batch row; deterministic fixed-order LDS reduction.
// ---------------------------------------------------------------------------
__global__ __launch_bounds__(256)
void head_kernel(const float* __restrict__ V, const float* __restrict__ Wc,
                 const float* __restrict__ bc, float* __restrict__ out) {
  __shared__ float part[256][NCLS + 1];
  __shared__ float logits[NCLS];
  const int t = threadIdx.x;
  const int b = blockIdx.x;

  float p[NCLS];
#pragma unroll
  for (int c = 0; c < NCLS; ++c) p[c] = 0.f;

  const float* vrow = V + (size_t)b * NCOLS;
  for (int f = t; f < NDIM; f += 256) {
    const float pv = (vrow[3 * f] + vrow[3 * f + 1] + vrow[3 * f + 2]) * (1.0f / 3.0f);
    const float* wr = Wc + (size_t)f * NCLS;
#pragma unroll
    for (int c = 0; c < NCLS; ++c) p[c] += pv * wr[c];
  }
#pragma unroll
  for (int c = 0; c < NCLS; ++c) part[t][c] = p[c];
  __syncthreads();

  if (t < NCLS) {
    float acc = bc[t];
    for (int i = 0; i < 256; ++i) acc += part[i][t];   // fixed order
    logits[t] = acc;
  }
  __syncthreads();

  if (t == 0) {
    float mx = logits[0];
    for (int c = 1; c < NCLS; ++c) mx = fmaxf(mx, logits[c]);
    float e[NCLS]; float sum = 0.f;
    for (int c = 0; c < NCLS; ++c) { e[c] = expf(logits[c] - mx); sum += e[c]; }
    const float inv = 1.f / sum;
    for (int c = 0; c < NCLS; ++c) out[(size_t)b * NCLS + c] = e[c] * inv;
  }
}

// ---------------------------------------------------------------------------
extern "C" void kernel_launch(void* const* d_in, const int* in_sizes, int n_in,
                              void* d_out, int out_size, void* d_ws, size_t ws_size,
                              hipStream_t stream) {
  const float* x0   = (const float*)d_in[0];   // x_dash
  const float* x1   = (const float*)d_in[1];   // x_rear
  const float* x2   = (const float*)d_in[2];   // x_right
  const float* Wnm0 = (const float*)d_in[3];   // Wd_nm
  const float* Wan0 = (const float*)d_in[4];   // Wd_an
  const float* Wnm1 = (const float*)d_in[5];   // Wr_nm
  const float* Wan1 = (const float*)d_in[6];   // Wr_an
  const float* Wnm2 = (const float*)d_in[7];   // Wt_nm
  const float* Wan2 = (const float*)d_in[8];   // Wt_an
  const float* Wc   = (const float*)d_in[9];
  const float* bc   = (const float*)d_in[10];

  // workspace layout:
  //   wsum : 3*8192 f32             @ byte 0
  //   S    : 3*128  f32             @ byte 98304
  //   Xb   : 3*128*8192 bf16        @ byte 99840  (16B aligned)
  //   V    : 128*6144 f32           @ byte 6391296 (16B aligned)
  float*  wsum = (float*)d_ws;
  float*  S    = wsum + 3 * KDIM;
  __bf16* Xb   = (__bf16*)((char*)d_ws + 99840);
  float*  V    = (float*)((char*)d_ws + 6391296);

  xcvt_kernel<<<1536, 256, 0, stream>>>(x0, x1, x2, Xb);
  rowsum_kernel<<<3 * KDIM / 8, 256, 0, stream>>>(Wan0, Wan1, Wan2, wsum);
  scale_kernel<<<(3 * BATCH) / 8, 256, 0, stream>>>(x0, x1, x2, wsum, S);

  const __bf16* Xb0 = Xb;
  const __bf16* Xb1 = Xb + (size_t)BATCH * KDIM;
  const __bf16* Xb2 = Xb + 2 * (size_t)BATCH * KDIM;
  gemm_scale_kernel<<<NDIM / BN, 256, 0, stream>>>(Xb0, Wnm0, S + 0 * BATCH, V, 0 * NDIM);
  gemm_scale_kernel<<<NDIM / BN, 256, 0, stream>>>(Xb1, Wnm1, S + 1 * BATCH, V, 1 * NDIM);
  gemm_scale_kernel<<<NDIM / BN, 256, 0, stream>>>(Xb2, Wnm2, S + 2 * BATCH, V, 2 * NDIM);

  head_kernel<<<BATCH, 256, 0, stream>>>(V, Wc, bc, (float*)d_out);
}
